// GODEVAE_r_43336220016766
// MI455X (gfx1250) — matrix operand order repaired and checked by reading, not compiled, not run on hardware
//
#include <hip/hip_runtime.h>
#include <hip/hip_bf16.h>
#include <math.h>

typedef __attribute__((ext_vector_type(16))) __bf16 v16bf;
typedef __attribute__((ext_vector_type(8)))  __bf16 v8bf;
typedef __attribute__((ext_vector_type(8)))  float  v8f;

// ---------------------------------------------------------------------------
// Utility kernels
// ---------------------------------------------------------------------------
__global__ void fill_f32_kernel(float* __restrict__ p, float v, int n) {
    int i = blockIdx.x * blockDim.x + threadIdx.x;
    if (i < n) p[i] = v;
}

__global__ void deg_kernel(const int* __restrict__ dst, float* __restrict__ deg, int e) {
    int i = blockIdx.x * blockDim.x + threadIdx.x;
    if (i < e) atomicAdd(&deg[dst[i]], 1.0f);
}

__global__ void rsqrt_kernel(const float* __restrict__ deg, float* __restrict__ dinv, int n) {
    int i = blockIdx.x * blockDim.x + threadIdx.x;
    if (i < n) {
        float d = deg[i];
        dinv[i] = (d > 0.f) ? rsqrtf(d) : 0.f;
    }
}

// ---------------------------------------------------------------------------
// Weight conversion: fp32 [K, C] -> bf16 swizzled into WMMA B-fragment order.
// Layout: [ks][col_tile][lane][16 elems] with, per ISA 7.12.2 (16-bit B 32x16):
//   lane = hl*16 + nl : column n = ct*16 + nl, K = ks*32 + hl*8 + (i&7) + (i>>3)*16
// so each lane's 16 bf16 (32 B) are contiguous -> 2x global_load_b128 in GEMM.
// ---------------------------------------------------------------------------
__global__ void cvt_w_swz_kernel(const float* __restrict__ W, __bf16* __restrict__ Bswz,
                                 int K, int C, int Ksteps) {
    const int TILES_N = C / 16;
    size_t total = (size_t)Ksteps * TILES_N * 32 * 16;
    size_t idx = (size_t)blockIdx.x * blockDim.x + threadIdx.x;
    if (idx >= total) return;
    int i    = (int)(idx & 15);         // element within lane fragment
    int lane = (int)((idx >> 4) & 31);
    size_t t = idx >> 9;                // ks * TILES_N + ct
    int ks = (int)(t / TILES_N);
    int ct = (int)(t % TILES_N);
    int hl = lane >> 4, nl = lane & 15;
    int p = i >> 1, w = i & 1;
    int k = ks * 32 + hl * 8 + (i & 7) + (i >> 3) * 16;
    int c = ct * 16 + nl;
    Bswz[idx] = (k < K) ? (__bf16)W[(size_t)k * C + c] : (__bf16)0.f;
}

// ---------------------------------------------------------------------------
// WMMA GEMM: C[M, COLS] = A[M, K](f32) @ W(bf16, pre-swizzled)
// 256 threads = 8 waves tiling a (ROWS x COLS) block.
// A tile staged fp32->bf16 into ping-pong LDS (one barrier per K-step);
// A fragment = 2x ds_load_b128, B fragment = 32 B contiguous global load.
// ---------------------------------------------------------------------------
template<int COLS>
__global__ __launch_bounds__(256) void gemm_bf16_wmma_kernel(
    const float* __restrict__ A, const __bf16* __restrict__ Bswz,
    float* __restrict__ C, int M, int K, int Ksteps)
{
    constexpr int TILES_N   = COLS / 16;     // 8 for COLS=128, 2 for COLS=32
    constexpr int ROW_TILES = 8 / TILES_N;   // 1 or 4
    constexpr int ROWS      = ROW_TILES * 16;
    constexpr int LSTR      = 40;            // bf16 elems/row: 80 B, 16 B aligned

    __shared__ __attribute__((aligned(16))) __bf16 As[2][ROWS * LSTR];

    const int tid  = threadIdx.x;
    const int wave = tid >> 5;
    const int lane = tid & 31;
    const int hl   = lane >> 4;
    const int ml   = lane & 15;
    const int col_tile = wave % TILES_N;
    const int row_tile = wave / TILES_N;
    const int rowBase  = blockIdx.x * ROWS;

    v8f acc = {0.f, 0.f, 0.f, 0.f, 0.f, 0.f, 0.f, 0.f};

    for (int ks = 0; ks < Ksteps; ++ks) {
        const int k0 = ks * 32;
        __bf16* buf = As[ks & 1];

        // Stage A tile (ROWS x 32) as packed bf16 pairs; unconditional loads
        // with clamped offset + value select (no exec-mask branches).
        #pragma unroll
        for (int it = 0; it < (ROWS * 32) / 512; ++it) {
            int idx = (it * 256 + tid) * 2;
            int r = idx >> 5, c = idx & 31;
            int row = rowBase + r, k = k0 + c;
            bool ok = (row < M) && (k < K);
            size_t off = ok ? ((size_t)row * K + k) : 0;
            float2 v = *reinterpret_cast<const float2*>(A + off);
            float vx = ok ? v.x : 0.f;
            float vy = ok ? v.y : 0.f;
            // Near-scope prefetch of next K tile (same rows, consumed by this
            // WGP next iteration): locality 3 -> WGP scope, all cache levels.
            if (ks + 1 < Ksteps) __builtin_prefetch(A + off + 32, 0, 3);
            union { __bf16 h[2]; unsigned u; } pk;
            pk.h[0] = (__bf16)vx; pk.h[1] = (__bf16)vy;
            *reinterpret_cast<unsigned*>(buf + r * LSTR + c) = pk.u;
        }
        __syncthreads();

        // A fragment (ISA 16-bit A 16x32): lane's 16 bf16 are two contiguous
        // 16 B runs: cols [hl*8 .. hl*8+7] and [16+hl*8 .. 16+hl*8+7].
        const __bf16* arow = buf + (row_tile * 16 + ml) * LSTR;
        v8bf alo = *reinterpret_cast<const v8bf*>(arow + hl * 8);
        v8bf ahi = *reinterpret_cast<const v8bf*>(arow + 16 + hl * 8);
        v16bf a = __builtin_shufflevector(alo, ahi,
                   0, 1, 2, 3, 4, 5, 6, 7, 8, 9, 10, 11, 12, 13, 14, 15);

        // B fragment: fully contiguous 32 B per lane (pre-swizzled).
        v16bf b = *reinterpret_cast<const v16bf*>(
            Bswz + (((size_t)ks * TILES_N + col_tile) * 32 + lane) * 16);

        acc = __builtin_amdgcn_wmma_f32_16x16x32_bf16(
            /*neg_a=*/false, a, /*neg_b=*/false, b,
            /*c_mod=*/(short)0, acc, /*reuse_a=*/false, /*reuse_b=*/false);
        // ping-pong buffers: next write goes to the other buffer; the single
        // barrier per iteration orders write(i+2) after all comp(i).
    }

    // C/D layout: VGPR r holds M = r + 8*hl, N = ml (per col tile).
    #pragma unroll
    for (int r = 0; r < 8; ++r) {
        int m = r + hl * 8;
        int row = rowBase + row_tile * 16 + m;
        if (row < M) C[(size_t)row * COLS + col_tile * 16 + ml] = acc[r];
    }
}

// ---------------------------------------------------------------------------
// GCN propagation: out = bias + self_loop + sum_edges
// ---------------------------------------------------------------------------
template<int F>
__global__ void init_agg_kernel(const float* __restrict__ hpre, const float* __restrict__ dinv,
                                const float* __restrict__ bias, float* __restrict__ out, int n) {
    size_t idx = (size_t)blockIdx.x * blockDim.x + threadIdx.x;
    if (idx >= (size_t)n * F) return;
    int i = (int)(idx / F);
    int f = (int)(idx % F);
    float di = dinv[i];
    out[idx] = bias[f] + hpre[idx] * di * di;   // self-loop norm = dinv^2
}

template<int F>
__global__ void scatter_add_kernel(const float* __restrict__ hpre,
                                   const int* __restrict__ src, const int* __restrict__ dst,
                                   const float* __restrict__ dinv, float* __restrict__ out, int e) {
    constexpr int G = F / 4;
    size_t gid = (size_t)blockIdx.x * blockDim.x + threadIdx.x;
    size_t ei = gid / G;
    int g = (int)(gid % G) * 4;
    if (ei >= (size_t)e) return;
    int s = src[ei], d = dst[ei];
    float w = dinv[s] * dinv[d];
    const float4 v = *reinterpret_cast<const float4*>(hpre + (size_t)s * F + g);
    float* o = out + (size_t)d * F + g;
    atomicAdd(o + 0, v.x * w);
    atomicAdd(o + 1, v.y * w);
    atomicAdd(o + 2, v.z * w);
    atomicAdd(o + 3, v.w * w);
}

// ---------------------------------------------------------------------------
// BatchNorm over nodes (feature dim = 128)
// ---------------------------------------------------------------------------
__global__ void bn_partial_kernel(const float* __restrict__ h,
                                  float* __restrict__ sums, float* __restrict__ sumsq, int n) {
    __shared__ float s1[256];
    __shared__ float s2[256];
    int tid = threadIdx.x;
    int f = tid & 127;
    int hh = tid >> 7;
    int r0 = blockIdx.x * 256 + hh;
    int rend = blockIdx.x * 256 + 256; if (rend > n) rend = n;
    float a = 0.f, b = 0.f;
    for (int r = r0; r < rend; r += 2) {
        float v = h[(size_t)r * 128 + f];
        a += v; b += v * v;
    }
    s1[tid] = a; s2[tid] = b;
    __syncthreads();
    if (hh == 0) {
        atomicAdd(&sums[f],  s1[tid] + s1[tid + 128]);
        atomicAdd(&sumsq[f], s2[tid] + s2[tid + 128]);
    }
}

__global__ void bn_final_kernel(const float* __restrict__ sums, const float* __restrict__ sumsq,
                                const float* __restrict__ g, const float* __restrict__ be,
                                float* __restrict__ scale, float* __restrict__ shift, int n) {
    int f = threadIdx.x;   // 128 threads
    float inv_n = 1.f / (float)n;
    float mu  = sums[f] * inv_n;
    float var = sumsq[f] * inv_n - mu * mu;
    float sc = g[f] * rsqrtf(var + 1e-5f);
    scale[f] = sc;
    shift[f] = be[f] - mu * sc;
}

// out = relu(agg*scale + shift) [+ residual]
__global__ void bn_apply_relu_kernel(const float* __restrict__ agg,
                                     const float* __restrict__ scale, const float* __restrict__ shift,
                                     const float* __restrict__ residual,
                                     float* __restrict__ out, size_t total) {
    size_t idx = (size_t)blockIdx.x * blockDim.x + threadIdx.x;
    if (idx >= total) return;
    int f = (int)(idx & 127);
    float v = fmaxf(agg[idx] * scale[f] + shift[f], 0.f);
    if (residual) v += residual[idx];
    out[idx] = v;
}

// ---------------------------------------------------------------------------
// t = sigmoid(h @ Wt + bt), Wt: [128]
// ---------------------------------------------------------------------------
__global__ void gate_kernel(const float* __restrict__ h, const float* __restrict__ Wt,
                            const float* __restrict__ bt, float* __restrict__ out, int n) {
    int i = blockIdx.x * blockDim.x + threadIdx.x;
    if (i >= n) return;
    float s = bt[0];
    #pragma unroll 8
    for (int f = 0; f < 128; ++f) s += h[(size_t)i * 128 + f] * Wt[f];
    out[i] = 1.f / (1.f + expf(-s));
}

// q_z = q_m + (softplus(q_s) + 1e-6) * eps
__global__ void reparam_kernel(const float* __restrict__ qm, const float* __restrict__ qs,
                               const float* __restrict__ eps, float* __restrict__ qz, size_t total) {
    size_t idx = (size_t)blockIdx.x * blockDim.x + threadIdx.x;
    if (idx >= total) return;
    float s = qs[idx];
    float sp = (s > 20.f) ? s : log1pf(expf(s));
    qz[idx] = qm[idx] + (sp + 1e-6f) * eps[idx];
}

// ---------------------------------------------------------------------------
// Launch
// ---------------------------------------------------------------------------
static inline int cdiv(long long a, int b) { return (int)((a + b - 1) / b); }

extern "C" void kernel_launch(void* const* d_in, const int* in_sizes, int n_in,
                              void* d_out, int out_size, void* d_ws, size_t ws_size,
                              hipStream_t stream) {
    constexpr int D_IN = 2000, H = 128, Z = 32;
    constexpr int K1PAD = 2016, K1STEPS = 63;   // ceil(2000/32)*32
    constexpr int K2STEPS = 4;                  // 128/32

    const int n = in_sizes[0] / D_IN;           // 50000
    const int e = in_sizes[1] / 2;              // 1600000

    const float* x   = (const float*)d_in[0];
    const int*   ei  = (const int*)  d_in[1];
    const int*   src = ei;
    const int*   dst = ei + e;
    const float* W1  = (const float*)d_in[2];
    const float* b1  = (const float*)d_in[3];
    const float* W2  = (const float*)d_in[4];
    const float* b2  = (const float*)d_in[5];
    const float* g1  = (const float*)d_in[6];
    const float* be1 = (const float*)d_in[7];
    const float* g2  = (const float*)d_in[8];
    const float* be2 = (const float*)d_in[9];
    const float* Wm  = (const float*)d_in[10];
    const float* bm  = (const float*)d_in[11];
    const float* Wv  = (const float*)d_in[12];
    const float* bv  = (const float*)d_in[13];
    const float* Wt  = (const float*)d_in[14];
    const float* bt  = (const float*)d_in[15];
    const float* eps = (const float*)d_in[16];

    float* out_qz = (float*)d_out;
    float* out_qm = out_qz + (size_t)n * Z;
    float* out_qs = out_qm + (size_t)n * Z;
    float* out_t  = out_qs + (size_t)n * Z;

    // Workspace carve-up
    float* ws    = (float*)d_ws;
    float* deg   = ws;                          // n
    float* dinv  = deg  + n;                    // n
    float* hpre  = dinv + n;                    // n*H
    float* hagg  = hpre + (size_t)n * H;        // n*H
    float* h1    = hagg + (size_t)n * H;        // n*H
    float* h2    = h1   + (size_t)n * H;        // n*H
    float* qpre  = h2   + (size_t)n * H;        // n*Z (reused for q_m and q_s)
    float* stats = qpre + (size_t)n * Z;        // 512: sums|sumsq|scale|shift
    float* sums = stats, * sumsq = stats + 128, * scale = stats + 256, * shift = stats + 384;
    __bf16* W1b = (__bf16*)(stats + 512);       // K1PAD*H   (= 63*8*512 swizzled)
    __bf16* W2b = W1b + (size_t)K1PAD * H;      // 128*H     (= 4*8*512)
    __bf16* Wmb = W2b + 128 * H;                // 128*Z     (= 4*2*512)
    __bf16* Wvb = Wmb + 128 * Z;                // 128*Z

    const int B = 256;

    // --- normalization: deg (with self-loop), dinv = rsqrt(deg) ---
    fill_f32_kernel<<<cdiv(n, B), B, 0, stream>>>(deg, 1.0f, n);
    deg_kernel<<<cdiv(e, B), B, 0, stream>>>(dst, deg, e);
    rsqrt_kernel<<<cdiv(n, B), B, 0, stream>>>(deg, dinv, n);

    // --- weight conversion to swizzled bf16 fragments ---
    cvt_w_swz_kernel<<<cdiv((long long)K1STEPS * 8 * 512, B), B, 0, stream>>>(W1, W1b, D_IN, H, K1STEPS);
    cvt_w_swz_kernel<<<cdiv((long long)K2STEPS * 8 * 512, B), B, 0, stream>>>(W2, W2b, H, H, K2STEPS);
    cvt_w_swz_kernel<<<cdiv((long long)K2STEPS * 2 * 512, B), B, 0, stream>>>(Wm, Wmb, H, Z, K2STEPS);
    cvt_w_swz_kernel<<<cdiv((long long)K2STEPS * 2 * 512, B), B, 0, stream>>>(Wv, Wvb, H, Z, K2STEPS);

    // --- layer 1: GEMM -> propagate -> BN -> ReLU ---
    gemm_bf16_wmma_kernel<128><<<cdiv(n, 16), B, 0, stream>>>(x, W1b, hpre, n, D_IN, K1STEPS);
    init_agg_kernel<128><<<cdiv((long long)n * H, B), B, 0, stream>>>(hpre, dinv, b1, hagg, n);
    scatter_add_kernel<128><<<cdiv((long long)e * 32, B), B, 0, stream>>>(hpre, src, dst, dinv, hagg, e);
    fill_f32_kernel<<<1, 256, 0, stream>>>(sums, 0.f, 256);   // zero sums+sumsq
    bn_partial_kernel<<<cdiv(n, 256), 256, 0, stream>>>(hagg, sums, sumsq, n);
    bn_final_kernel<<<1, 128, 0, stream>>>(sums, sumsq, g1, be1, scale, shift, n);
    bn_apply_relu_kernel<<<cdiv((long long)n * H, B), B, 0, stream>>>(hagg, scale, shift, nullptr, h1, (size_t)n * H);

    // --- layer 2: GEMM -> propagate -> BN -> ReLU -> +residual ---
    gemm_bf16_wmma_kernel<128><<<cdiv(n, 16), B, 0, stream>>>(h1, W2b, hpre, n, H, K2STEPS);
    init_agg_kernel<128><<<cdiv((long long)n * H, B), B, 0, stream>>>(hpre, dinv, b2, hagg, n);
    scatter_add_kernel<128><<<cdiv((long long)e * 32, B), B, 0, stream>>>(hpre, src, dst, dinv, hagg, e);
    fill_f32_kernel<<<1, 256, 0, stream>>>(sums, 0.f, 256);
    bn_partial_kernel<<<cdiv(n, 256), 256, 0, stream>>>(hagg, sums, sumsq, n);
    bn_final_kernel<<<1, 128, 0, stream>>>(sums, sumsq, g2, be2, scale, shift, n);
    bn_apply_relu_kernel<<<cdiv((long long)n * H, B), B, 0, stream>>>(hagg, scale, shift, h1, h2, (size_t)n * H);

    // --- gate t ---
    gate_kernel<<<cdiv(n, B), B, 0, stream>>>(h2, Wt, bt, out_t, n);

    // --- q_m head: GEMM (COLS=32) -> propagate into d_out ---
    gemm_bf16_wmma_kernel<32><<<cdiv(n, 64), B, 0, stream>>>(h2, Wmb, qpre, n, H, K2STEPS);
    init_agg_kernel<32><<<cdiv((long long)n * Z, B), B, 0, stream>>>(qpre, dinv, bm, out_qm, n);
    scatter_add_kernel<32><<<cdiv((long long)e * 8, B), B, 0, stream>>>(qpre, src, dst, dinv, out_qm, e);

    // --- q_s head ---
    gemm_bf16_wmma_kernel<32><<<cdiv(n, 64), B, 0, stream>>>(h2, Wvb, qpre, n, H, K2STEPS);
    init_agg_kernel<32><<<cdiv((long long)n * Z, B), B, 0, stream>>>(qpre, dinv, bv, out_qs, n);
    scatter_add_kernel<32><<<cdiv((long long)e * 8, B), B, 0, stream>>>(qpre, src, dst, dinv, out_qs, e);

    // --- reparameterization ---
    reparam_kernel<<<cdiv((long long)n * Z, B), B, 0, stream>>>(out_qm, out_qs, eps, out_qz, (size_t)n * Z);
}
